// PCELayer_1039382086112
// MI455X (gfx1250) — compile-verified, matrix operand order checked by reading, not run on hardware
//
#include <hip/hip_runtime.h>
#include <hip/hip_bf16.h>
#include <math.h>
#include <stdint.h>

// ---------------------------------------------------------------------------
// Problem constants (fixed by the reference)
// ---------------------------------------------------------------------------
#define B_    8
#define C_    128        // CIN == COUT == 128
#define H_    128
#define W_    128
#define HW_   16384      // H_*W_
#define G_    8          // groupnorm groups
#define E_    8          // experts
#define KCH   36         // K chunks of 32 (9 taps * 4 cin-blocks)
#define NGRP  262144     // elements per GN group = (C/G)*H*W
#define EPS_  1e-5f

typedef __attribute__((ext_vector_type(16))) _Float16 v16h;
typedef __attribute__((ext_vector_type(8)))  _Float16 v8h;
typedef __attribute__((ext_vector_type(8)))  float    v8f;

union V16U { v16h v; v8h h[2]; };

__device__ __forceinline__ float silu_f(float x) {
    return x / (1.0f + expf(-x));
}

// gfx1250 async global->LDS copy (16B per lane), tracked by ASYNCcnt
__device__ __forceinline__ void async_copy_b128(uint32_t lds_off, const void* gptr) {
    asm volatile("global_load_async_to_lds_b128 %0, %1, off"
                 :: "v"(lds_off), "v"(gptr) : "memory");
}
__device__ __forceinline__ void wait_asynccnt0() {
    asm volatile("s_wait_asynccnt 0" ::: "memory");
}

// ---------------------------------------------------------------------------
// Zero a small stats region (graph-capture-safe alternative to memset)
// ---------------------------------------------------------------------------
__global__ void zero_stats_kernel(float* p) {
    p[threadIdx.x] = 0.0f;   // 128 threads: gsum[64] + gsq[64]
}

// ---------------------------------------------------------------------------
// f32 -> f16 activation pre-convert (done once, not 9x inside the conv loop)
// ---------------------------------------------------------------------------
__global__ void to_f16_kernel(const float* __restrict__ src,
                              _Float16* __restrict__ dst) {
    size_t idx = (size_t)blockIdx.x * 256 + threadIdx.x;
    dst[idx] = (_Float16)src[idx];
}

// ---------------------------------------------------------------------------
// Router: softmax(fourier_feats @ router_w + b) over 8x8 patches, 8 experts
// ---------------------------------------------------------------------------
__global__ void router_kernel(const float* __restrict__ rw,   // [16][8]
                              const float* __restrict__ rb,   // [8]
                              float* __restrict__ wts)        // [8][8][8]
{
    int t = threadIdx.x;
    if (t >= 64) return;
    int py = t >> 3, px = t & 7;
    float cy = (py + 0.5f) / 8.0f;
    float cx = (px + 0.5f) / 8.0f;
    float feats[16];
#pragma unroll
    for (int f = 0; f < 4; ++f) {
        float fr = exp2f((float)f) * 3.14159265358979323846f;
        feats[f]      = sinf(cy * fr);
        feats[4 + f]  = cosf(cy * fr);
        feats[8 + f]  = sinf(cx * fr);
        feats[12 + f] = cosf(cx * fr);
    }
    float logit[8];
    float mx = -1e30f;
#pragma unroll
    for (int e = 0; e < 8; ++e) {
        float s = rb[e];
#pragma unroll
        for (int k = 0; k < 16; ++k) s += feats[k] * rw[k * 8 + e];
        logit[e] = s;
        mx = fmaxf(mx, s);
    }
    float denom = 0.0f;
#pragma unroll
    for (int e = 0; e < 8; ++e) { logit[e] = expf(logit[e] - mx); denom += logit[e]; }
    float rinv = 1.0f / denom;
#pragma unroll
    for (int e = 0; e < 8; ++e) wts[t * 8 + e] = logit[e] * rinv;
}

// ---------------------------------------------------------------------------
// Pack conv weights [128][128][3][3] f32 -> f16 tiles [36 kchunk][128 m][32 k]
// K ordering: k = tap*128 + cin (tap = ky*3+kx), chunk = (tap, cin/32)
// ---------------------------------------------------------------------------
__global__ void pack_weights_kernel(const float* __restrict__ w,
                                    _Float16* __restrict__ wp)
{
    int idx = blockIdx.x * 256 + threadIdx.x;      // 36*128*32 = 147456
    if (idx >= KCH * 128 * 32) return;
    int kl  = idx & 31;
    int m   = (idx >> 5) & 127;
    int kc  = idx >> 12;
    int tap = kc >> 2;
    int cb  = kc & 3;
    int cin = cb * 32 + kl;
    wp[idx] = (_Float16)w[(m * 128 + cin) * 9 + tap];
}

// ---------------------------------------------------------------------------
// Implicit-GEMM conv3x3 (SAME) via v_wmma_f32_16x16x32_f16.
// Block = 256 threads (8 waves) = one image row (128 px) x all 128 out chans.
// K loop: ky(3, uniform boundary skip) x cinblock(4). Per (ky,cb):
//   - A: 3 dx tap panels (8KB each, pre-packed f16) copied global->LDS with
//        async b128 (per-lane dests scatter into bank-conflict-padded rows)
//   - B: one 130-row halo panel (f16 activations, vector b128 loads),
//        reused by all 3 dx taps as a +/-1 LDS row shift -> no divergence
//   - compute: 3 x 8 WMMA (24 per stage), 288 per wave total
// Wave w owns M rows [16w,16w+16) == GN group w -> wave-local stat reduce.
// ---------------------------------------------------------------------------
__global__ __launch_bounds__(256)
void conv3x3_wmma_kernel(const _Float16* __restrict__ xh,   // [B][128][16384] f16
                         const _Float16* __restrict__ wp,   // [36][128][32] f16
                         float*          __restrict__ outbuf,// [B][128][16384] f32
                         float*          __restrict__ gsum,  // [B*8]
                         float*          __restrict__ gsq)   // [B*8]
{
    const int h    = blockIdx.x;    // output row 0..127
    const int b    = blockIdx.y;    // 0..7
    const int tid  = threadIdx.x;
    const int wave = tid >> 5;
    const int lane = tid & 31;

    __shared__ __align__(16) _Float16 Alds[3][128][40];  // 80B rows -> aligned frags
    __shared__ __align__(16) _Float16 Blds[130][40];     // rows 0,129 = zero halo

    // zero halo rows once (only columns 0..31 are ever read)
    if (tid < 64) {
        int r = (tid >> 5) ? 129 : 0;
        Blds[r][lane] = (_Float16)0.0f;
    }

    v8f acc[8];
#pragma unroll
    for (int i = 0; i < 8; ++i)
#pragma unroll
        for (int r = 0; r < 8; ++r) acc[i][r] = 0.0f;

    const _Float16* xb = xh + (size_t)b * C_ * HW_;
    const uint32_t abase = (uint32_t)(uintptr_t)&Alds[0][0][0];

    for (int ky = 0; ky < 3; ++ky) {
        const int hs = h + ky - 1;
        if (hs < 0 || hs >= H_) continue;          // block-uniform skip
        for (int cb = 0; cb < 4; ++cb) {
            __syncthreads();
            // ---- stage A: 3 tap panels, async global->LDS, pad-aware scatter
#pragma unroll
            for (int dxi = 0; dxi < 3; ++dxi) {
                const char* src = (const char*)(wp + (size_t)((ky * 3 + dxi) * 4 + cb) * 4096);
#pragma unroll
                for (int half = 0; half < 2; ++half) {
                    const int uu  = tid + half * 256;   // b128 unit id (512 total)
                    const int m   = uu >> 2;
                    const int seg = uu & 3;
                    async_copy_b128(abase + (uint32_t)(dxi * 10240 + m * 80 + seg * 16),
                                    src + uu * 16);
                }
            }
            // ---- stage B: halo panel, vector b128 global loads of f16 rows
#pragma unroll
            for (int it = 0; it < 2; ++it) {
                const int cin  = (tid >> 4) + 16 * it;  // 0..31
                const int span = tid & 15;              // 8-pixel span
                const _Float16* src = xb + (size_t)(cb * 32 + cin) * HW_
                                         + hs * W_ + span * 8;
                v8h vv = *(const v8h*)src;
#pragma unroll
                for (int i = 0; i < 8; ++i) Blds[span * 8 + i + 1][cin] = vv[i];
            }
            wait_asynccnt0();
            __syncthreads();
            // ---- compute: 3 dx taps x 8 N-chunks of WMMA
            const int mlo = lane & 15;
            const int kb  = (lane >> 4) * 8;
#pragma unroll
            for (int dxi = 0; dxi < 3; ++dxi) {
                V16U ua;
                ua.h[0] = *(const v8h*)&Alds[dxi][wave * 16 + mlo][kb];
                ua.h[1] = *(const v8h*)&Alds[dxi][wave * 16 + mlo][kb + 16];
#pragma unroll
                for (int nc = 0; nc < 8; ++nc) {
                    // output px n needs input w = n + dxi - 1 -> LDS row n + dxi
                    const int brow = nc * 16 + mlo + dxi;
                    V16U ub;
                    ub.h[0] = *(const v8h*)&Blds[brow][kb];
                    ub.h[1] = *(const v8h*)&Blds[brow][kb + 16];
                    acc[nc] = __builtin_amdgcn_wmma_f32_16x16x32_f16(
                        false, ua.v, false, ub.v, (short)0, acc[nc], false, false);
                }
            }
        }
    }

    // ---- writeback + GN statistics (group g == wave)
    float s = 0.0f, sq = 0.0f;
    const int mbase = wave * 16 + 8 * (lane >> 4);
    const int nlo   = lane & 15;
#pragma unroll
    for (int nc = 0; nc < 8; ++nc) {
        const int p = h * 128 + nc * 16 + nlo;
#pragma unroll
        for (int r = 0; r < 8; ++r) {
            const float v = acc[nc][r];
            outbuf[((size_t)b * C_ + (mbase + r)) * HW_ + p] = v;
            s  += v;
            sq += v * v;
        }
    }
#pragma unroll
    for (int off = 16; off > 0; off >>= 1) {
        s  += __shfl_down(s,  off, 32);
        sq += __shfl_down(sq, off, 32);
    }
    if (lane == 0) {
        atomicAdd(&gsum[b * G_ + wave], s);
        atomicAdd(&gsq [b * G_ + wave], sq);
    }
}

// ---------------------------------------------------------------------------
// Finalize GN stats: mean + rsqrt(var+eps)
// ---------------------------------------------------------------------------
__global__ void finalize_stats_kernel(const float* __restrict__ gsum,
                                      const float* __restrict__ gsq,
                                      float* __restrict__ mean,
                                      float* __restrict__ inv)
{
    int t = threadIdx.x;
    if (t >= B_ * G_) return;
    const float invN = 1.0f / (float)NGRP;
    float m = gsum[t] * invN;
    float v = gsq[t] * invN - m * m;
    mean[t] = m;
    inv[t]  = rsqrtf(v + EPS_);
}

// ---------------------------------------------------------------------------
// Fused expert epilogue: silu(GN(conv)) + x, router-weighted accumulate
// ---------------------------------------------------------------------------
__global__ void merge_acc_kernel(const float* __restrict__ conv,
                                 const float* __restrict__ xin,
                                 const float* __restrict__ eg,   // [128] (per-expert)
                                 const float* __restrict__ eb,   // [128]
                                 const float* __restrict__ mean, // [B*8]
                                 const float* __restrict__ inv,  // [B*8]
                                 const float* __restrict__ wts,  // [8][8][8]
                                 int e,
                                 float* __restrict__ merged)
{
    size_t idx = (size_t)blockIdx.x * 256 + threadIdx.x;   // 16,777,216 total
    int p  = (int)(idx & 16383);
    int c  = (int)((idx >> 14) & 127);
    int bg = (int)(idx >> 21) * G_ + (c >> 4);
    float v  = conv[idx];
    float xn = (v - mean[bg]) * inv[bg];
    float hh = xn * eg[c] + eb[c];
    float o  = silu_f(hh) + xin[idx];
    int hp = p >> 11;            // patch row
    int wq = (p & 127) >> 4;     // patch col
    float wt = wts[(hp * 8 + wq) * 8 + e];
    float prev = (e == 0) ? 0.0f : merged[idx];
    merged[idx] = prev + wt * o;
}

// ---------------------------------------------------------------------------
// GN stats over [B][128][HW] buffer (for merged tensor), grid = (32, B*G)
// ---------------------------------------------------------------------------
__global__ void gn_stats_kernel(const float* __restrict__ buf,
                                float* __restrict__ gsum,
                                float* __restrict__ gsq)
{
    int bg = blockIdx.y;
    int b = bg >> 3, g = bg & 7;
    const float* base = buf + ((size_t)b * C_ + g * 16) * HW_;
    float s = 0.0f, q = 0.0f;
    for (size_t i = (size_t)blockIdx.x * 256 + threadIdx.x; i < (size_t)NGRP;
         i += (size_t)32 * 256) {
        float v = base[i];
        s += v;
        q += v * v;
    }
#pragma unroll
    for (int off = 16; off > 0; off >>= 1) {
        s += __shfl_down(s, off, 32);
        q += __shfl_down(q, off, 32);
    }
    __shared__ float ss[8], sqh[8];
    int wv = threadIdx.x >> 5, ln = threadIdx.x & 31;
    if (ln == 0) { ss[wv] = s; sqh[wv] = q; }
    __syncthreads();
    if (threadIdx.x == 0) {
        float S = 0.0f, Q = 0.0f;
#pragma unroll
        for (int i = 0; i < 8; ++i) { S += ss[i]; Q += sqh[i]; }
        atomicAdd(&gsum[bg], S);
        atomicAdd(&gsq[bg], Q);
    }
}

// ---------------------------------------------------------------------------
// y = x + gamma * silu(GN(merged))  -> written directly as f16 for post conv
// ---------------------------------------------------------------------------
__global__ void y_kernel(const float* __restrict__ merged,
                         const float* __restrict__ xin,
                         const float* __restrict__ mg,
                         const float* __restrict__ mb,
                         const float* __restrict__ mean,
                         const float* __restrict__ inv,
                         const float* __restrict__ gamma,
                         _Float16* __restrict__ yh)
{
    size_t idx = (size_t)blockIdx.x * 256 + threadIdx.x;
    int c  = (int)((idx >> 14) & 127);
    int bg = (int)(idx >> 21) * G_ + (c >> 4);
    float xn = (merged[idx] - mean[bg]) * inv[bg];
    float hh = xn * mg[c] + mb[c];
    yh[idx] = (_Float16)(xin[idx] + gamma[0] * silu_f(hh));
}

// ---------------------------------------------------------------------------
// out = silu(GN(post_conv))
// ---------------------------------------------------------------------------
__global__ void out_kernel(const float* __restrict__ pc,
                           const float* __restrict__ pg,
                           const float* __restrict__ pb,
                           const float* __restrict__ mean,
                           const float* __restrict__ inv,
                           float* __restrict__ out)
{
    size_t idx = (size_t)blockIdx.x * 256 + threadIdx.x;
    int c  = (int)((idx >> 14) & 127);
    int bg = (int)(idx >> 21) * G_ + (c >> 4);
    float xn = (pc[idx] - mean[bg]) * inv[bg];
    float hh = xn * pg[c] + pb[c];
    out[idx] = silu_f(hh);
}

// ---------------------------------------------------------------------------
// Host orchestration
// ---------------------------------------------------------------------------
extern "C" void kernel_launch(void* const* d_in, const int* in_sizes, int n_in,
                              void* d_out, int out_size, void* d_ws, size_t ws_size,
                              hipStream_t stream)
{
    (void)in_sizes; (void)n_in; (void)out_size; (void)ws_size;

    const float* x        = (const float*)d_in[0];   // [8][128][128][128]
    const float* expert_w = (const float*)d_in[1];   // [8][128][128][3][3]
    const float* expert_g = (const float*)d_in[2];   // [8][128]
    const float* expert_b = (const float*)d_in[3];   // [8][128]
    const float* router_w = (const float*)d_in[4];   // [16][8]
    const float* router_b = (const float*)d_in[5];   // [8]
    const float* merge_g  = (const float*)d_in[6];   // [128]
    const float* merge_b  = (const float*)d_in[7];   // [128]
    const float* gamma    = (const float*)d_in[8];   // [1]
    const float* post_w   = (const float*)d_in[9];   // [128][128][3][3]
    const float* post_g   = (const float*)d_in[10];  // [128]
    const float* post_b   = (const float*)d_in[11];  // [128]
    float* out = (float*)d_out;

    char* ws = (char*)d_ws;
    float*     wts   = (float*)(ws);                        // 512 floats
    float*     gsum  = (float*)(ws + 4096);                 // 64
    float*     gsq   = gsum + 64;                           // 64
    float*     gmean = gsum + 128;                          // 64
    float*     ginv  = gsum + 192;                          // 64
    _Float16*  wpack = (_Float16*)(ws + 8192);              // 9 * 147456 halves
    _Float16*  acth  = (_Float16*)(ws + (size_t)(4  << 20));// 32 MB: xh, later yh
    float*     bufA  = (float*)(ws + (size_t)(40 << 20));   // 64 MB (conv out)
    float*     bufB  = (float*)(ws + (size_t)(104 << 20));  // 64 MB (merged/post)

    const size_t WPK = (size_t)KCH * 128 * 32;              // 147456 halves/set

    // Router weights for the 8x8 patch grid
    router_kernel<<<1, 64, 0, stream>>>(router_w, router_b, wts);

    // Pack all weight sets to f16 implicit-GEMM layout (8 experts + post)
    for (int e = 0; e < E_; ++e)
        pack_weights_kernel<<<576, 256, 0, stream>>>(
            expert_w + (size_t)e * C_ * C_ * 9, wpack + (size_t)e * WPK);
    pack_weights_kernel<<<576, 256, 0, stream>>>(post_w, wpack + (size_t)E_ * WPK);

    const dim3 cgrid(128, B_);
    const int  EW = (B_ * C_ * HW_) / 256;   // 65536 blocks, elementwise passes

    // Pre-convert activations to f16 once
    to_f16_kernel<<<EW, 256, 0, stream>>>(x, acth);

    // Sequential experts: conv(+stats) -> finalize -> fused GN/SiLU/res/merge
    for (int e = 0; e < E_; ++e) {
        zero_stats_kernel<<<1, 128, 0, stream>>>(gsum);
        conv3x3_wmma_kernel<<<cgrid, 256, 0, stream>>>(
            acth, wpack + (size_t)e * WPK, bufA, gsum, gsq);
        finalize_stats_kernel<<<1, 64, 0, stream>>>(gsum, gsq, gmean, ginv);
        merge_acc_kernel<<<EW, 256, 0, stream>>>(
            bufA, x, expert_g + e * C_, expert_b + e * C_,
            gmean, ginv, wts, e, bufB);
    }

    // merged GN -> y = x + gamma*silu(GN(merged)), written as f16 (acth reused)
    zero_stats_kernel<<<1, 128, 0, stream>>>(gsum);
    gn_stats_kernel<<<dim3(32, B_ * G_), 256, 0, stream>>>(bufB, gsum, gsq);
    finalize_stats_kernel<<<1, 64, 0, stream>>>(gsum, gsq, gmean, ginv);
    y_kernel<<<EW, 256, 0, stream>>>(bufB, x, merge_g, merge_b, gmean, ginv,
                                     gamma, acth);

    // post conv (same WMMA kernel) -> GN -> SiLU
    zero_stats_kernel<<<1, 128, 0, stream>>>(gsum);
    conv3x3_wmma_kernel<<<cgrid, 256, 0, stream>>>(
        acth, wpack + (size_t)E_ * WPK, bufB, gsum, gsq);
    finalize_stats_kernel<<<1, 64, 0, stream>>>(gsum, gsq, gmean, ginv);
    out_kernel<<<EW, 256, 0, stream>>>(bufB, post_g, post_b, gmean, ginv, out);
}
